// SlidingWindowAttention_87711822119333
// MI455X (gfx1250) — matrix operand order repaired and checked by reading, not compiled, not run on hardware
//
#include <hip/hip_runtime.h>
#include <hip/hip_bf16.h>
#include <math.h>

// ---------------------------------------------------------------------------
// Sliding-window attention for MI455X (gfx1250), wave32, WMMA f16->f32.
//   1) qkv_gemm : x @ {Wq,Wk,Wv} + bias -> q,k,v  (f16, [B,H,S,64] layout)
//   2) attn     : flash-style banded softmax(QK^T/8)V -> attn_out f16 [B,S,1024]
//   3) oproj    : attn_out @ Wo + bo -> f32 out
// Software-pipelined: global loads for tile i+1 are issued before the WMMA
// compute phase of tile i, hiding HBM latency behind matrix work.
// Workspace: 4 f16 buffers of 4M halves each = 32 MB total.
// ---------------------------------------------------------------------------

typedef __attribute__((ext_vector_type(16))) _Float16 v16h;
typedef __attribute__((ext_vector_type(8)))  _Float16 v8h;
typedef __attribute__((ext_vector_type(8)))  float    v8f;

constexpr int SEQ = 2048;
constexpr int DM  = 1024;
constexpr int NH  = 16;
constexpr int DK  = 64;
constexpr int HWIN = 256;
constexpr float NEGINF = -1e30f;

__device__ __forceinline__ v16h make_frag(const _Float16* p0, const _Float16* p1) {
    union { v16h v; v8h h[2]; } u;
    u.h[0] = *(const v8h*)p0;
    u.h[1] = *(const v8h*)p1;
    return u.v;
}

// A-operand (16x32, M = lane&15):
//   lanes 0-15 : K 0..15 ; lanes 16-31: K 16..31
__device__ __forceinline__ v16h loadA(const _Float16* row, int hi) {
    return make_frag(row + hi * 16, row + hi * 16 + 8);
}

// B-operand (32x16, col N = lane&15, LDS stored [N][K] so K contiguous):
//   lanes 0-15: K 0..15 ; lanes 16-31: K 16..31
__device__ __forceinline__ v16h loadB(const _Float16* rowN_kbase) {
    return make_frag(rowN_kbase, rowN_kbase + 8);
}

// ---------------------------------------------------------------------------
// Kernel 1: fused QKV projection.  grid = (64 Mtiles, 16 Ntiles, 3), 128 thr.
// ---------------------------------------------------------------------------
__global__ __launch_bounds__(128) void qkv_gemm(
    const float* __restrict__ x,
    const float* __restrict__ Wq, const float* __restrict__ bq,
    const float* __restrict__ Wk, const float* __restrict__ bk,
    const float* __restrict__ Wv, const float* __restrict__ bv,
    _Float16* __restrict__ qh, _Float16* __restrict__ kh,
    _Float16* __restrict__ vh)
{
    const int mBase = blockIdx.x * 64;
    const int nBase = blockIdx.y * 64;
    const int which = blockIdx.z;
    const float* W    = (which == 0) ? Wq : (which == 1) ? Wk : Wv;
    const float* bias = (which == 0) ? bq : (which == 1) ? bk : bv;
    _Float16*   outp  = (which == 0) ? qh : (which == 1) ? kh : vh;

    __shared__ __align__(16) _Float16 As[64][40];  // [m][k]
    __shared__ __align__(16) _Float16 Bs[64][40];  // [n][k] (K-transposed)

    const int tid  = threadIdx.x;
    const int wave = tid >> 5;
    const int lane = tid & 31;
    const int hi   = lane >> 4;
    const int ln   = lane & 15;

    const int r = tid >> 1;              // staged row (A) / col (B)
    const int c = (tid & 1) * 16;        // staged k-offset

    float xa[16], wa[16];                // pipeline registers (next tile, f32)
    auto loadTiles = [&](int k0) {
        const float4* sx = (const float4*)(x + (size_t)(mBase + r) * DM + k0 + c);
#pragma unroll
        for (int i = 0; i < 4; ++i) {
            float4 f = sx[i];
            xa[4*i+0] = f.x; xa[4*i+1] = f.y; xa[4*i+2] = f.z; xa[4*i+3] = f.w;
        }
        const float* sw = W + (size_t)(k0 + c) * DM + nBase + r;
#pragma unroll
        for (int i = 0; i < 16; ++i) wa[i] = sw[(size_t)i * DM];
    };

    v8f acc[4] = {};
    loadTiles(0);

    for (int k0 = 0; k0 < DM; k0 += 32) {
        __syncthreads();
        {   // stage current tile (registers -> f16 -> LDS)
            _Float16 ta[16], tb[16];
#pragma unroll
            for (int i = 0; i < 16; ++i) { ta[i] = (_Float16)xa[i]; tb[i] = (_Float16)wa[i]; }
            *(v8h*)&As[r][c]     = *(v8h*)&ta[0];
            *(v8h*)&As[r][c + 8] = *(v8h*)&ta[8];
            *(v8h*)&Bs[r][c]     = *(v8h*)&tb[0];
            *(v8h*)&Bs[r][c + 8] = *(v8h*)&tb[8];
        }
        __syncthreads();

        if (k0 + 32 < DM) loadTiles(k0 + 32);   // prefetch next tile into regs

        const v16h a = loadA(&As[16 * wave + ln][0], hi);
        v16h b4[4];
#pragma unroll
        for (int t = 0; t < 4; ++t) b4[t] = loadB(&Bs[16 * t + ln][hi * 16]);
#pragma unroll
        for (int t = 0; t < 4; ++t)
            acc[t] = __builtin_amdgcn_wmma_f32_16x16x32_f16(
                false, a, false, b4[t], (short)0, acc[t], false, false);
    }

    // C layout: (m,n) -> VGPR r=m%8, lane=(m/8)*16+n
#pragma unroll
    for (int t = 0; t < 4; ++t) {
        const int col = nBase + 16 * t + ln;
        const float bc = bias[col];
        const int h = col >> 6;
        const int d = col & 63;
#pragma unroll
        for (int rr = 0; rr < 8; ++rr) {
            const int row = mBase + 16 * wave + rr + 8 * hi;
            const int b = row >> 11;
            const int s = row & (SEQ - 1);
            outp[(((size_t)(b * NH + h)) * SEQ + s) * DK + d] =
                (_Float16)(acc[t][rr] + bc);
        }
    }
}

// ---------------------------------------------------------------------------
// Kernel 2: banded flash attention. grid = (B*H, SEQ/64), 128 threads.
// ---------------------------------------------------------------------------
__global__ __launch_bounds__(128) void attn_kernel(
    const _Float16* __restrict__ qh, const _Float16* __restrict__ kh,
    const _Float16* __restrict__ vh, _Float16* __restrict__ ah)
{
    const int bh = blockIdx.x;               // b*16 + h
    const int qb = blockIdx.y * 64;
    const int tid  = threadIdx.x;
    const int wave = tid >> 5;
    const int lane = tid & 31;
    const int hi   = lane >> 4;
    const int ln   = lane & 15;

    const _Float16* Qp = qh + (size_t)bh * SEQ * DK;
    const _Float16* Kp = kh + (size_t)bh * SEQ * DK;
    const _Float16* Vp = vh + (size_t)bh * SEQ * DK;

    __shared__ __align__(16) _Float16 Kbuf[32][72];      // [key][d]
    __shared__ __align__(16) _Float16 Vbuf[64][40];      // [d][key] (transposed)
    __shared__ __align__(16) _Float16 Pbuf[4][16][40];   // per-wave P tile

    // Q fragments for this wave's 16 queries (d 0..31 and 32..63)
    const _Float16* qrow = Qp + (size_t)(qb + 16 * wave + ln) * DK;
    const v16h qa0 = loadA(qrow, hi);
    const v16h qa1 = loadA(qrow + 32, hi);

    // K/V staging (pipelined in registers)
    const int kKey = tid >> 2;               // K: key row, 16-half chunk
    const int kC   = (tid & 3) * 16;
    const int vD   = tid >> 1;               // V: d row (transposed store)
    const int vC   = (tid & 1) * 16;
    _Float16 kreg[16], vreg[16];
    auto loadKV = [&](int kt) {
        const int gk = kt + kKey;
        if (gk < SEQ) {
            const v8h* src = (const v8h*)(Kp + (size_t)gk * DK + kC);
            *(v8h*)&kreg[0] = src[0];
            *(v8h*)&kreg[8] = src[1];
        } else {
            v8h z = {};
            *(v8h*)&kreg[0] = z; *(v8h*)&kreg[8] = z;
        }
#pragma unroll
        for (int i = 0; i < 16; ++i) {
            const int gv = kt + vC + i;
            vreg[i] = (gv < SEQ) ? Vp[(size_t)gv * DK + vD] : (_Float16)0.0f;
        }
    };

    v8f  o[4] = {};
    float rmax[8], rsum[8];
#pragma unroll
    for (int rr = 0; rr < 8; ++rr) { rmax[rr] = NEGINF; rsum[rr] = 0.0f; }

    int ktStart = qb - HWIN; if (ktStart < 0) ktStart = 0;   // 32-aligned
    int ktEnd   = qb + 64 + HWIN; if (ktEnd > SEQ) ktEnd = SEQ;

    loadKV(ktStart);

    for (int kt = ktStart; kt < ktEnd; kt += 32) {
        __syncthreads();
        // stage current K/V tile from registers into LDS
        *(v8h*)&Kbuf[kKey][kC]     = *(v8h*)&kreg[0];
        *(v8h*)&Kbuf[kKey][kC + 8] = *(v8h*)&kreg[8];
        *(v8h*)&Vbuf[vD][vC]       = *(v8h*)&vreg[0];
        *(v8h*)&Vbuf[vD][vC + 8]   = *(v8h*)&vreg[8];
        __syncthreads();

        if (kt + 32 < ktEnd) loadKV(kt + 32);   // prefetch next tile into regs

        // scores S = Q K^T : all 4 B-fragments first, then 4 WMMAs
        v16h kb[4];
        kb[0] = loadB(&Kbuf[ln][hi * 16]);
        kb[1] = loadB(&Kbuf[ln][32 + hi * 16]);
        kb[2] = loadB(&Kbuf[16 + ln][hi * 16]);
        kb[3] = loadB(&Kbuf[16 + ln][32 + hi * 16]);
        v8f s[2] = {};
        s[0] = __builtin_amdgcn_wmma_f32_16x16x32_f16(false, qa0, false, kb[0], (short)0, s[0], false, false);
        s[0] = __builtin_amdgcn_wmma_f32_16x16x32_f16(false, qa1, false, kb[1], (short)0, s[0], false, false);
        s[1] = __builtin_amdgcn_wmma_f32_16x16x32_f16(false, qa0, false, kb[2], (short)0, s[1], false, false);
        s[1] = __builtin_amdgcn_wmma_f32_16x16x32_f16(false, qa1, false, kb[3], (short)0, s[1], false, false);

        // scale + band mask (large-negative sentinel keeps exp() NaN-free)
        const float scale = 0.125f;   // 1/sqrt(64)
#pragma unroll
        for (int nt = 0; nt < 2; ++nt) {
            const int kidx = kt + 16 * nt + ln;
#pragma unroll
            for (int rr = 0; rr < 8; ++rr) {
                const int q = qb + 16 * wave + rr + 8 * hi;
                int dlt = q - kidx; if (dlt < 0) dlt = -dlt;
                const bool ok = (kidx < SEQ) && (dlt <= HWIN);
                s[nt][rr] = ok ? s[nt][rr] * scale : NEGINF;
            }
        }

        // online softmax update
#pragma unroll
        for (int rr = 0; rr < 8; ++rr) {
            float m = fmaxf(s[0][rr], s[1][rr]);
            m = fmaxf(m, __shfl_xor(m, 1));
            m = fmaxf(m, __shfl_xor(m, 2));
            m = fmaxf(m, __shfl_xor(m, 4));
            m = fmaxf(m, __shfl_xor(m, 8));
            const float nm    = fmaxf(rmax[rr], m);
            const float alpha = __expf(rmax[rr] - nm);
            rmax[rr] = nm;

            const float p0 = __expf(s[0][rr] - nm);
            const float p1 = __expf(s[1][rr] - nm);
            Pbuf[wave][rr + 8 * hi][ln]      = (_Float16)p0;
            Pbuf[wave][rr + 8 * hi][16 + ln] = (_Float16)p1;

            float ps = p0 + p1;
            ps += __shfl_xor(ps, 1);
            ps += __shfl_xor(ps, 2);
            ps += __shfl_xor(ps, 4);
            ps += __shfl_xor(ps, 8);
            rsum[rr] = rsum[rr] * alpha + ps;
#pragma unroll
            for (int t = 0; t < 4; ++t) o[t][rr] *= alpha;
        }
        __syncthreads();

        // O += P @ V   (P re-read from LDS in A layout; all frags hoisted)
        const v16h pa = loadA(&Pbuf[wave][ln][0], hi);
        v16h vb[4];
#pragma unroll
        for (int t = 0; t < 4; ++t) vb[t] = loadB(&Vbuf[16 * t + ln][hi * 16]);
#pragma unroll
        for (int t = 0; t < 4; ++t)
            o[t] = __builtin_amdgcn_wmma_f32_16x16x32_f16(
                false, pa, false, vb[t], (short)0, o[t], false, false);
    }

    // write attn_out as f16 [B, S, H*64]
    const int b = bh >> 4;
    const int h = bh & 15;
#pragma unroll
    for (int t = 0; t < 4; ++t) {
#pragma unroll
        for (int rr = 0; rr < 8; ++rr) {
            const int q = qb + 16 * wave + rr + 8 * hi;
            const float val = o[t][rr] / rsum[rr];
            ah[((size_t)(b * SEQ + q)) * DM + h * DK + 16 * t + ln] = (_Float16)val;
        }
    }
}

// ---------------------------------------------------------------------------
// Kernel 3: output projection attn_out(f16) @ Wo + bo -> f32 out.
// ---------------------------------------------------------------------------
__global__ __launch_bounds__(128) void oproj_gemm(
    const _Float16* __restrict__ ah, const float* __restrict__ Wo,
    const float* __restrict__ bo, float* __restrict__ out)
{
    const int mBase = blockIdx.x * 64;
    const int nBase = blockIdx.y * 64;

    __shared__ __align__(16) _Float16 As[64][40];
    __shared__ __align__(16) _Float16 Bs[64][40];

    const int tid  = threadIdx.x;
    const int wave = tid >> 5;
    const int lane = tid & 31;
    const int hi   = lane >> 4;
    const int ln   = lane & 15;

    const int r = tid >> 1;
    const int c = (tid & 1) * 16;

    v8h  aa[2];                 // pipeline regs: A tile chunk (f16)
    float wa[16];               // pipeline regs: Wo tile chunk (f32)
    auto loadTiles = [&](int k0) {
        const v8h* sa = (const v8h*)(ah + (size_t)(mBase + r) * DM + k0 + c);
        aa[0] = sa[0]; aa[1] = sa[1];
        const float* sw = Wo + (size_t)(k0 + c) * DM + nBase + r;
#pragma unroll
        for (int i = 0; i < 16; ++i) wa[i] = sw[(size_t)i * DM];
    };

    v8f acc[4] = {};
    loadTiles(0);

    for (int k0 = 0; k0 < DM; k0 += 32) {
        __syncthreads();
        {
            _Float16 tb[16];
#pragma unroll
            for (int i = 0; i < 16; ++i) tb[i] = (_Float16)wa[i];
            *(v8h*)&As[r][c]     = aa[0];
            *(v8h*)&As[r][c + 8] = aa[1];
            *(v8h*)&Bs[r][c]     = *(v8h*)&tb[0];
            *(v8h*)&Bs[r][c + 8] = *(v8h*)&tb[8];
        }
        __syncthreads();

        if (k0 + 32 < DM) loadTiles(k0 + 32);

        const v16h a = loadA(&As[16 * wave + ln][0], hi);
        v16h b4[4];
#pragma unroll
        for (int t = 0; t < 4; ++t) b4[t] = loadB(&Bs[16 * t + ln][hi * 16]);
#pragma unroll
        for (int t = 0; t < 4; ++t)
            acc[t] = __builtin_amdgcn_wmma_f32_16x16x32_f16(
                false, a, false, b4[t], (short)0, acc[t], false, false);
    }

#pragma unroll
    for (int t = 0; t < 4; ++t) {
        const int col = nBase + 16 * t + ln;
        const float bc = bo[col];
#pragma unroll
        for (int rr = 0; rr < 8; ++rr) {
            const int row = mBase + 16 * wave + rr + 8 * hi;
            out[(size_t)row * DM + col] = acc[t][rr] + bc;
        }
    }
}

// ---------------------------------------------------------------------------
extern "C" void kernel_launch(void* const* d_in, const int* in_sizes, int n_in,
                              void* d_out, int out_size, void* d_ws, size_t ws_size,
                              hipStream_t stream) {
    const float* x  = (const float*)d_in[0];
    const float* Wq = (const float*)d_in[1];
    const float* bq = (const float*)d_in[2];
    const float* Wk = (const float*)d_in[3];
    const float* bk = (const float*)d_in[4];
    const float* Wv = (const float*)d_in[5];
    const float* bv = (const float*)d_in[6];
    const float* Wo = (const float*)d_in[7];
    const float* bo = (const float*)d_in[8];
    float* out = (float*)d_out;

    // workspace: 4 x (B*H*S*DK = 4,194,304) f16 buffers = 32 MB
    _Float16* ws = (_Float16*)d_ws;
    const size_t per = (size_t)2 * NH * SEQ * DK;
    _Float16* qh = ws;
    _Float16* kh = ws + per;
    _Float16* vh = ws + 2 * per;
    _Float16* ah = ws + 3 * per;

    dim3 blk(128);
    qkv_gemm   <<<dim3(64, 16, 3),        blk, 0, stream>>>(x, Wq, bq, Wk, bk, Wv, bv, qh, kh, vh);
    attn_kernel<<<dim3(2 * NH, SEQ / 64), blk, 0, stream>>>(qh, kh, vh, ah);
    oproj_gemm <<<dim3(64, 16),           blk, 0, stream>>>(ah, Wo, bo, out);
}
